// ActionNet_30150670418004
// MI455X (gfx1250) — matrix operand order repaired and checked by reading, not compiled, not run on hardware
//
#include <hip/hip_runtime.h>
#include <hip/hip_bf16.h>

typedef __attribute__((ext_vector_type(16))) __bf16 v16bf;
typedef __attribute__((ext_vector_type(8)))  float  v8f;

__device__ __forceinline__ unsigned short f2bf_bits(float f) {
    unsigned u = __float_as_uint(f);
    unsigned r = u + 0x7FFFu + ((u >> 16) & 1u);   // round-to-nearest-even
    return (unsigned short)(r >> 16);
}
__device__ __forceinline__ float sigm(float x) { return 1.f / (1.f + __expf(-x)); }

// ---------------------------------------------------------------------------
// Convert conv weights f32 [Co, Ci*27] -> bf16 [Co, Kp] (zero padded K->Kp)
// ---------------------------------------------------------------------------
__global__ void wconv_bf16(const float* __restrict__ w, unsigned short* __restrict__ wb,
                           int Co, int K, int Kp) {
    int i = blockIdx.x * 256 + threadIdx.x;
    if (i >= Co * Kp) return;
    int co = i / Kp, k = i % Kp;
    float f = (k < K) ? w[(size_t)co * K + k] : 0.f;
    wb[i] = f2bf_bits(f);
}

// ---------------------------------------------------------------------------
// Convert activations f32 [NC,D,H,W] -> bf16 zero-padded [NC,D+2,H+2,W+2].
// Removes all boundary predicates and cvt math from the WMMA inner loop.
// ---------------------------------------------------------------------------
__global__ void xpad_bf16(const float* __restrict__ x, unsigned short* __restrict__ xp,
                          int NC, int D, int H, int W) {
    int pD = D + 2, pH = H + 2, pW = W + 2;
    long total = (long)NC * pD * pH * pW;
    long i = (long)blockIdx.x * 256 + threadIdx.x;
    if (i >= total) return;
    int ww = (int)(i % pW); long t = i / pW;
    int hh = (int)(t % pH); t /= pH;
    int dd = (int)(t % pD); int nc = (int)(t / pD);
    int d = dd - 1, h = hh - 1, w = ww - 1;
    float f = 0.f;
    if ((unsigned)d < (unsigned)D && (unsigned)h < (unsigned)H && (unsigned)w < (unsigned)W)
        f = x[((size_t)nc * D + d) * H * W + h * W + w];
    xp[i] = f2bf_bits(f);
}

// ---------------------------------------------------------------------------
// Mask field: m0[p] = softmax(1x1x1 conv -> 2ch)[0] per spatial position
// ---------------------------------------------------------------------------
__global__ void mask_field(const float* __restrict__ x, const float* __restrict__ mw,
                           const float* __restrict__ mb, float* __restrict__ m0,
                           int N, int Ci, int DHW) {
    int p = blockIdx.x * 256 + threadIdx.x;
    if (p >= N * DHW) return;
    int n = p / DHW, sp = p % DHW;
    float s0 = mb[0], s1 = mb[1];
    const float* xp = x + (size_t)n * Ci * DHW + sp;
    for (int ci = 0; ci < Ci; ++ci) {
        float v = xp[(size_t)ci * DHW];
        s0 += mw[ci] * v;
        s1 += mw[Ci + ci] * v;
    }
    m0[p] = 1.f / (1.f + __expf(s1 - s0));
}

// ---------------------------------------------------------------------------
// Implicit-GEMM masked conv3d (k=3, pad=1) via V_WMMA_F32_16X16X32_BF16.
// A = weights bf16 [Co,Kp]; B gathered from padded bf16 activations using a
// precomputed LDS offset table (no div/mod, no bounds checks in loop).
// Block = 4 waves; each wave: one 16-co tile x two 16-pos tiles (2 acc chains,
// shared A fragment). y layout: [Co, P].
// ---------------------------------------------------------------------------
__global__ __launch_bounds__(128) void conv3d_wmma(
    const unsigned short* __restrict__ xp, const unsigned short* __restrict__ wb,
    const float* __restrict__ cb, const float* __restrict__ m0,
    float* __restrict__ y, int N, int Ci, int Co, int D, int H, int W, int Kp) {
    const int K = Ci * 27;
    const int HW = H * W, DHW = D * HW, P = N * DHW;
    const int pW = W + 2, pH = H + 2, pD = D + 2;
    const int pHW = pH * pW, pDHW = pD * pHW;

    __shared__ int offT[6944];                 // up to Kp=6912 (Ci=256) + slack

    int tid = threadIdx.x, lane = tid & 31, wave = tid >> 5;
    for (int k = tid; k < Kp; k += 128) {      // one-time k -> padded-offset decode
        int kc = (k < K) ? k : (K - 1);
        int ci = kc / 27, t = kc % 27;
        offT[k] = ci * pDHW + (t / 9) * pHW + ((t % 9) / 3) * pW + (t % 3);
    }
    __syncthreads();

    int coBase = blockIdx.x * 16;
    int pT = blockIdx.y * 128 + wave * 32;
    int ncol = lane & 15;
    int p0 = pT + ncol, p1 = p0 + 16;
    int pc0 = (p0 < P) ? p0 : (P - 1);
    int pc1 = (p1 < P) ? p1 : (P - 1);
    int n0 = pc0 / DHW, sp0 = pc0 % DHW;
    int n1 = pc1 / DHW, sp1 = pc1 % DHW;
    int d0 = sp0 / HW, h0 = (sp0 % HW) / W, w0 = sp0 % W;
    int d1 = sp1 / HW, h1 = (sp1 % HW) / W, w1 = sp1 % W;
    size_t b0 = ((size_t)(n0 * Ci) * pD + d0) * pHW + h0 * pW + w0;
    size_t b1 = ((size_t)(n1 * Ci) * pD + d1) * pHW + h1 * pW + w1;

    int khalfA = (lane < 16) ? 0 : 8;          // ISA 16-bit A-fragment layout
    int khalfB = (lane < 16) ? 0 : 16;         // unused; B gather uses khalfA (same k order as A)
    const unsigned* wrow =
        (const unsigned*)wb + ((size_t)(coBase + (lane & 15)) * Kp >> 1);

    v8f acc0 = {}, acc1 = {};
    for (int k0 = 0; k0 < K; k0 += 32) {
        union { v16bf v; unsigned u[8]; } A, B0, B1;
#pragma unroll
        for (int vg = 0; vg < 8; ++vg) {
            int k = k0 + ((vg >> 2) << 4) + ((vg & 3) << 1) + khalfA;  // even
            A.u[vg] = wrow[k >> 1];
        }
#pragma unroll
        for (int vg = 0; vg < 8; ++vg) {
            int kk = ((vg >> 2) << 4) + ((vg & 3) << 1) + khalfA;
            int o0 = offT[k0 + kk], o1 = offT[k0 + kk + 1];
            unsigned lo0 = xp[b0 + o0], hi0 = xp[b0 + o1];
            unsigned lo1 = xp[b1 + o0], hi1 = xp[b1 + o1];
            B0.u[vg] = lo0 | (hi0 << 16);
            B1.u[vg] = lo1 | (hi1 << 16);
        }
        acc0 = __builtin_amdgcn_wmma_f32_16x16x32_bf16(false, A.v, false, B0.v,
                                                       (short)0, acc0, false, false);
        acc1 = __builtin_amdgcn_wmma_f32_16x16x32_bf16(false, A.v, false, B1.v,
                                                       (short)0, acc1, false, false);
    }

    float mm0 = m0[pc0], mm1 = m0[pc1];
#pragma unroll
    for (int vg = 0; vg < 8; ++vg) {           // C layout: VGPR vg -> M = vg (+8 hi lanes)
        int co = coBase + vg + khalfA;
        float bias = cb[co];
        if (p0 < P) y[(size_t)co * P + p0] = (acc0[vg] + bias) * mm0;
        if (p1 < P) y[(size_t)co * P + p1] = (acc1[vg] + bias) * mm1;
    }
}

// ---------------------------------------------------------------------------
// Training-mode BN statistics per channel over all N*D*H*W positions.
// ---------------------------------------------------------------------------
__global__ __launch_bounds__(256) void bn_stats(const float* __restrict__ y,
                                                float* __restrict__ mean,
                                                float* __restrict__ rstd, int P) {
    int c = blockIdx.x;
    const float* yc = y + (size_t)c * P;
    float s = 0.f, s2 = 0.f;
    for (int i = threadIdx.x; i < P; i += 256) { float v = yc[i]; s += v; s2 += v * v; }
    __shared__ float sh[256], sh2[256];
    sh[threadIdx.x] = s; sh2[threadIdx.x] = s2;
    __syncthreads();
    for (int st = 128; st > 0; st >>= 1) {
        if (threadIdx.x < st) { sh[threadIdx.x] += sh[threadIdx.x + st];
                                sh2[threadIdx.x] += sh2[threadIdx.x + st]; }
        __syncthreads();
    }
    if (threadIdx.x == 0) {
        float mn = sh[0] / P;
        float var = sh2[0] / P - mn * mn;
        mean[c] = mn;
        rstd[c] = rsqrtf(var + 1e-5f);
    }
}

// ---------------------------------------------------------------------------
// Fused BN-apply + ReLU + maxpool.  y:[C, N*DHW] -> xout:[N,C,Do,Ho,Wo]
// ---------------------------------------------------------------------------
__global__ void bn_relu_pool(const float* __restrict__ y, const float* __restrict__ mean,
                             const float* __restrict__ rstd, const float* __restrict__ gam,
                             const float* __restrict__ bet, float* __restrict__ xout,
                             int N, int C, int D, int H, int W, int pd, int ph, int pw) {
    int Do = D / pd, Ho = H / ph, Wo = W / pw;
    long total = (long)N * C * Do * Ho * Wo;
    long i = (long)blockIdx.x * 256 + threadIdx.x;
    if (i >= total) return;
    int wo = (int)(i % Wo); long t = i / Wo;
    int ho = (int)(t % Ho); t /= Ho;
    int dq = (int)(t % Do); t /= Do;
    int c  = (int)(t % C);  int n = (int)(t / C);
    int HW = H * W, DHW = D * HW;
    float sc = rstd[c] * gam[c];
    float sb = bet[c] - mean[c] * sc;
    const float* yc = y + (size_t)c * ((size_t)N * DHW) + (size_t)n * DHW;
    float mx = 0.f;                                   // ReLU outputs are >= 0
    for (int id = 0; id < pd; ++id)
        for (int ih = 0; ih < ph; ++ih)
            for (int iw = 0; iw < pw; ++iw) {
                float e = yc[(dq * pd + id) * HW + (ho * ph + ih) * W + (wo * pw + iw)];
                e = fmaxf(e * sc + sb, 0.f);
                mx = fmaxf(mx, e);
            }
    xout[(((size_t)(n * C + c) * Do + dq) * Ho + ho) * Wo + wo] = mx;
}

// ---------------------------------------------------------------------------
// Head: bilinear-attention LSTM scan (T=4, H=512) + semantic + classifier.
// One workgroup (16 waves), everything staged in LDS.
// ---------------------------------------------------------------------------
__global__ __launch_bounds__(512) void head_kernel(
    const float* __restrict__ xfea,   // [4,256,45]
    const float* __restrict__ g,      // [4,256]
    const float* __restrict__ attW, const float* __restrict__ attb,
    const float* __restrict__ semW, const float* __restrict__ semb,
    const float* __restrict__ Wih,  const float* __restrict__ Whh,
    const float* __restrict__ bih,  const float* __restrict__ bhh,
    const float* __restrict__ c1W,  const float* __restrict__ c1b,
    const float* __restrict__ c2W,  const float* __restrict__ c2b,
    float* __restrict__ out) {
    const int T = 4, C = 256, NP = 45;
    __shared__ float h[512], cs[512], vv[256], ipt[256], sc[48];
    __shared__ float gate[2048], yy[512], red[512], hss[2048];
    int tid = threadIdx.x;
    h[tid] = 0.f; cs[tid] = 0.f;
    __syncthreads();

    for (int t = 0; t < T; ++t) {
        if (tid < 256) {                          // v = attW @ h
            float s = 0.f; const float* wr = attW + (size_t)tid * 512;
            for (int j = 0; j < 512; ++j) s += wr[j] * h[j];
            vv[tid] = s;
        }
        __syncthreads();
        const float* ft = xfea + (size_t)t * C * NP;
        if (tid < NP) {                           // scores[n] = feat[n]·v + attb
            float s = attb[0];
            for (int c2 = 0; c2 < C; ++c2) s += ft[(size_t)c2 * NP + tid] * vv[c2];
            sc[tid] = s;
        }
        __syncthreads();
        if (tid == 0) {                           // softmax over 45 positions
            float mx = sc[0];
            for (int i = 1; i < NP; ++i) mx = fmaxf(mx, sc[i]);
            float sm = 0.f;
            for (int i = 0; i < NP; ++i) { float e = __expf(sc[i] - mx); sc[i] = e; sm += e; }
            float inv = 1.f / sm;
            for (int i = 0; i < NP; ++i) sc[i] *= inv;
        }
        __syncthreads();
        if (tid < 256) {                          // ipt[c] = sum_n a[n] feat[n,c]
            float s = 0.f;
            for (int nn = 0; nn < NP; ++nn) s += sc[nn] * ft[(size_t)tid * NP + nn];
            ipt[tid] = s;
        }
        __syncthreads();
        for (int i = tid; i < 2048; i += 512) {   // gates = Wih@ipt + Whh@h + biases
            float s = bih[i] + bhh[i];
            const float* wi = Wih + (size_t)i * 256;
            for (int c2 = 0; c2 < 256; ++c2) s += wi[c2] * ipt[c2];
            const float* wh = Whh + (size_t)i * 512;
            for (int j = 0; j < 512; ++j) s += wh[j] * h[j];
            gate[i] = s;
        }
        __syncthreads();
        {                                         // LSTM cell update (i,f,g,o order)
            float ig = gate[tid], fg = gate[512 + tid];
            float gg = gate[1024 + tid], og = gate[1536 + tid];
            float cn = sigm(fg) * cs[tid] + sigm(ig) * tanhf(gg);
            float hn = sigm(og) * tanhf(cn);
            cs[tid] = cn; h[tid] = hn; hss[t * 512 + tid] = hn;
        }
        __syncthreads();
    }

    for (int t = 0; t < T; ++t) {
        const float* ht = hss + t * 512;
        {                                         // se = normalize(semW@h + semb)
            float s = semb[tid];
            const float* wr = semW + (size_t)tid * 512;
            for (int j = 0; j < 512; ++j) s += wr[j] * ht[j];
            yy[tid] = s; red[tid] = s * s;
        }
        __syncthreads();
        for (int st = 256; st > 0; st >>= 1) {
            if (tid < st) red[tid] += red[tid + st];
            __syncthreads();
        }
        float nrm = fmaxf(sqrtf(red[0]), 1e-12f);
        out[240 + t * 512 + tid] = yy[tid] / nrm;
        __syncthreads();
        {                                         // y = relu(c1W @ [h; g] + c1b)
            float s2 = c1b[tid];
            const float* w1 = c1W + (size_t)tid * 768;
            for (int j = 0; j < 512; ++j) s2 += w1[j] * ht[j];
            const float* gt = g + t * 256;
            for (int j = 0; j < 256; ++j) s2 += w1[512 + j] * gt[j];
            yy[tid] = fmaxf(s2, 0.f);
        }
        __syncthreads();
        if (tid < 60) {                           // logits = c2W @ y + c2b
            float s3 = c2b[tid];
            const float* w2 = c2W + (size_t)tid * 512;
            for (int j = 0; j < 512; ++j) s3 += w2[j] * yy[j];
            out[t * 60 + tid] = s3;
        }
        __syncthreads();
    }
}

// ---------------------------------------------------------------------------
// Host orchestration
// d_in: 0=data, 1+6i..6+6i = stage-i {cw,cb,mw,mb,g,b}, 25..30 = g-branch,
// 31=attW 32=attb 33=semW 34=semb 35=Wih 36=Whh 37=bih 38=bhh 39=c1W 40=c1b
// 41=c2W 42=c2b.  d_out: logits[1,4,60] ++ se[1,4,512] (2288 f32).
// ---------------------------------------------------------------------------
extern "C" void kernel_launch(void* const* d_in, const int* in_sizes, int n_in,
                              void* d_out, int out_size, void* d_ws, size_t ws_size,
                              hipStream_t stream) {
    (void)in_sizes; (void)n_in; (void)out_size; (void)ws_size;
    float* ws   = (float*)d_ws;
    float* bufA = ws;                                     //  6M floats (act ping)
    float* bufB = ws + (size_t)6  * 1024 * 1024;          //  6M floats (act pong)
    float* Y    = ws + (size_t)12 * 1024 * 1024;          // 48M floats (conv out [Co,P])
    float* M0   = ws + (size_t)60 * 1024 * 1024;          //  1M floats (mask field)
    unsigned short* WB = (unsigned short*)(ws + (size_t)61 * 1024 * 1024); // 4M u16
    unsigned short* XP = (unsigned short*)(ws + (size_t)63 * 1024 * 1024); // 16M u16
    float* STAT = ws + (size_t)71 * 1024 * 1024;          // mean[256], rstd[256]
    float* G    = STAT + 512;                             // [4,256]

    static const int dims[5][3]  = {{48,48,80},{24,24,40},{12,12,20},{6,6,10},{3,3,5}};
    static const int chans[5][2] = {{1,64},{64,128},{128,256},{256,256},{256,256}};
    float* outs[4] = {bufA, bufB, bufA, bufB};

    const float* cur = (const float*)d_in[0];             // data == [4,1,48,48,80]
    for (int s = 0; s < 5; ++s) {
        int Ci = chans[s][0], Co = chans[s][1];
        int D = dims[s][0], H = dims[s][1], W = dims[s][2];
        int DHW = D * H * W, P = 4 * DHW;
        int K = Ci * 27, Kp = (K + 31) & ~31;
        int base = (s < 4) ? (1 + 6 * s) : 25;
        const float* cw = (const float*)d_in[base + 0];
        const float* cb = (const float*)d_in[base + 1];
        const float* mw = (const float*)d_in[base + 2];
        const float* mb = (const float*)d_in[base + 3];
        const float* gg = (const float*)d_in[base + 4];
        const float* bb = (const float*)d_in[base + 5];

        int wtot = Co * Kp;
        wconv_bf16<<<(wtot + 255) / 256, 256, 0, stream>>>(cw, WB, Co, K, Kp);

        long ptot = (long)4 * Ci * (D + 2) * (H + 2) * (W + 2);
        xpad_bf16<<<(int)((ptot + 255) / 256), 256, 0, stream>>>(cur, XP, 4 * Ci, D, H, W);

        mask_field<<<(P + 255) / 256, 256, 0, stream>>>(cur, mw, mb, M0, 4, Ci, DHW);

        dim3 cg(Co / 16, (P + 127) / 128);
        conv3d_wmma<<<cg, 128, 0, stream>>>(XP, WB, cb, M0, Y, 4, Ci, Co, D, H, W, Kp);

        bn_stats<<<Co, 256, 0, stream>>>(Y, STAT, STAT + 256, P);

        int pd, ph, pw; float* xo;
        if (s < 4) { pd = ph = pw = 2; xo = outs[s]; }
        else       { pd = 3; ph = 3; pw = 5; xo = G; }
        long tot = (long)4 * Co * (D / pd) * (H / ph) * (W / pw);
        bn_relu_pool<<<(int)((tot + 255) / 256), 256, 0, stream>>>(
            Y, STAT, STAT + 256, gg, bb, xo, 4, Co, D, H, W, pd, ph, pw);
        if (s < 4) cur = outs[s];                          // after s=3: bufB = x_fea
    }

    head_kernel<<<1, 512, 0, stream>>>(
        bufB, G,
        (const float*)d_in[31], (const float*)d_in[32],
        (const float*)d_in[33], (const float*)d_in[34],
        (const float*)d_in[35], (const float*)d_in[36],
        (const float*)d_in[37], (const float*)d_in[38],
        (const float*)d_in[39], (const float*)d_in[40],
        (const float*)d_in[41], (const float*)d_in[42],
        (float*)d_out);
}
